// DonutSwinLayer_17497696764144
// MI455X (gfx1250) — matrix operand order repaired and checked by reading, not compiled, not run on hardware
//
#include <hip/hip_runtime.h>
#include <hip/hip_bf16.h>
#include <stdint.h>

// ---------------- types ----------------
typedef __attribute__((ext_vector_type(16))) __bf16 v16bf;
typedef __attribute__((ext_vector_type(8)))  float  v8f;
typedef __attribute__((ext_vector_type(4)))  unsigned int v4u;
typedef __attribute__((ext_vector_type(8)))  int    v8i;
typedef __attribute__((ext_vector_type(4)))  int    v4i;

#if defined(__has_builtin)
#if __has_builtin(__builtin_amdgcn_tensor_load_to_lds) && __has_builtin(__builtin_amdgcn_s_wait_tensorcnt)
#define HAVE_TDM 1
#endif
#endif

// ---------------- problem constants ----------------
#define BB     16
#define CC     512
#define NHEAD  16
#define HDIM   32
#define WSZ    8
#define SSH    4
#define NTOK   64
#define MLPD   2048
#define NWIN   1024
#define ROWS   65536

// ---------------- workspace layout (ushort units) ----------------
#define OFF_WQ  ((size_t)0)
#define OFF_WK  ((size_t)262144)
#define OFF_WV  ((size_t)524288)
#define OFF_WP  ((size_t)786432)
#define OFF_W1  ((size_t)1048576)
#define OFF_W2  ((size_t)2097152)
#define OFF_XW  ((size_t)3145728)
#define OFF_Q   (OFF_XW  + (size_t)ROWS * CC)
#define OFF_K   (OFF_Q   + (size_t)ROWS * CC)
#define OFF_V   (OFF_K   + (size_t)ROWS * CC)
#define OFF_CTX (OFF_V   + (size_t)ROWS * CC)
#define OFF_XLN OFF_CTX                    // overlays ctx (dead after proj)
#define OFF_Y1  OFF_XW                     // overlays Xw/Q/K/V (dead after attn)

// ---------------- helpers ----------------
__device__ __forceinline__ unsigned short f2b(float f) {
    unsigned int u = __builtin_bit_cast(unsigned int, f);
    u += 0x7FFFu + ((u >> 16) & 1u);       // RNE
    return (unsigned short)(u >> 16);
}
__device__ __forceinline__ __bf16 us2bf(unsigned short u) {
    return __builtin_bit_cast(__bf16, u);
}
__device__ __forceinline__ v8f wmma_bf16(v16bf a, v16bf b, v8f c) {
    return __builtin_amdgcn_wmma_f32_16x16x32_bf16(false, a, false, b,
                                                   (short)0, c, false, false);
}

// A fragment: 16x32 bf16, row-major source, leading dim ld (elements).
__device__ __forceinline__ v16bf load_a_frag(const unsigned short* p, int ld) {
    int lane = threadIdx.x & 31;
    const unsigned short* rp = p + (size_t)(lane & 15) * ld;
    int kb = (lane >> 4) * 16;
    v16bf a;
#pragma unroll
    for (int v = 0; v < 8; ++v) {
        int k = kb + 2 * v;
        a[2 * v]     = us2bf(rp[k]);
        a[2 * v + 1] = us2bf(rp[k + 1]);
    }
    return a;
}
// B fragment: 32x16, element (k,n) at p[n*ld + k]
__device__ __forceinline__ v16bf load_b_colk(const unsigned short* p, int ld) {
    int lane = threadIdx.x & 31;
    const unsigned short* cp = p + (size_t)(lane & 15) * ld + ((lane >> 4) * 16);
    v16bf b;
#pragma unroll
    for (int j = 0; j < 16; ++j) b[j] = us2bf(cp[j]);
    return b;
}
// B fragment: 32x16, element (k,n) at p[k*ld + n]
__device__ __forceinline__ v16bf load_b_rowk(const unsigned short* p, int ld) {
    int lane = threadIdx.x & 31;
    const unsigned short* cp = p + (size_t)((lane >> 4) * 16) * ld + (lane & 15);
    v16bf b;
#pragma unroll
    for (int j = 0; j < 16; ++j) b[j] = us2bf(cp[(size_t)j * ld]);
    return b;
}

#ifdef HAVE_TDM
// 2D Tensor-DMA: DMA a (tile_d1 x tile_d0) bf16 tile (row stride = stride_d0
// elements) from global to LDS (packed row-major). D# per CDNA5 ISA 8.3/8.4.
// 6-arg builtin: (v4u g0, v8i g1, v4i g2, v4i g3, v8i g4, i32 cpol)
__device__ __forceinline__ void tdm_load_2d(unsigned lds_off, const void* gaddr,
                                            unsigned tile_d0, unsigned tile_d1,
                                            unsigned stride_d0) {
    unsigned long long ga = (unsigned long long)(size_t)gaddr;
    v4u g0;
    g0[0] = 1u;                                              // count=1, user D#
    g0[1] = lds_off;                                         // lds_addr
    g0[2] = (unsigned)(ga & 0xFFFFFFFFu);                    // global_addr lo
    g0[3] = (unsigned)((ga >> 32) & 0x1FFFFFFu) | (2u << 30);// addr hi | type=2
    v8i g1;
    g1[0] = (int)(1u << 16);                                 // data_size=2B
    g1[1] = (int)((tile_d0 & 0xFFFFu) << 16);                // tensor_dim0 lo16
    g1[2] = (int)((tile_d1 & 0xFFFFu) << 16);                // dim0 hi=0 | tensor_dim1 lo16
    g1[3] = (int)(((tile_d1 >> 16) & 0xFFFFu) | ((tile_d0 & 0xFFFFu) << 16)); // dim1 hi | tile_dim0
    g1[4] = (int)(tile_d1 & 0xFFFFu);                        // tile_dim1
    g1[5] = (int)stride_d0;                                  // tensor_dim0_stride lo32
    g1[6] = 0;
    g1[7] = 0;
    v4i z4 = {0, 0, 0, 0};
    v8i z8 = {0, 0, 0, 0, 0, 0, 0, 0};
    __builtin_amdgcn_tensor_load_to_lds(g0, g1, z4, z4, z8, 0);
}
#endif

// ---------------- weight fp32 -> bf16 ----------------
__global__ void k_cvt(const float* __restrict__ src, unsigned short* __restrict__ dst, int n) {
    int i = blockIdx.x * 256 + threadIdx.x;
    if (i < n) dst[i] = f2b(src[i]);
}

// ---------------- LN1 + cyclic shift + window partition -> bf16 ----------------
__global__ __launch_bounds__(256)
void k_prep(const float* __restrict__ hs, const float* __restrict__ g,
            const float* __restrict__ be, unsigned short* __restrict__ xw) {
    int row  = blockIdx.x * 8 + (threadIdx.x >> 5);
    int lane = threadIdx.x & 31;
    int win = row >> 6, tok = row & 63;
    int b = win >> 6, wy = (win >> 3) & 7, wx = win & 7;
    int iy = tok >> 3, ix = tok & 7;
    int hh = (wy * WSZ + iy + SSH) & 63;
    int ww = (wx * WSZ + ix + SSH) & 63;
    const float* src = hs + (((size_t)b * 4096) + (size_t)hh * 64 + ww) * CC;
    float x[16], s = 0.f, s2 = 0.f;
#pragma unroll
    for (int i = 0; i < 16; ++i) { float v = src[lane + i * 32]; x[i] = v; s += v; s2 += v * v; }
#pragma unroll
    for (int m = 1; m <= 16; m <<= 1) { s += __shfl_xor(s, m, 32); s2 += __shfl_xor(s2, m, 32); }
    float mu = s * (1.f / CC);
    float inv = rsqrtf(s2 * (1.f / CC) - mu * mu + 1e-5f);
    unsigned short* dst = xw + (size_t)row * CC;
#pragma unroll
    for (int i = 0; i < 16; ++i) {
        int c = lane + i * 32;
        dst[c] = f2b((x[i] - mu) * inv * g[c] + be[c]);
    }
}

// ---------------- LN2 (natural row order) -> bf16 ----------------
__global__ __launch_bounds__(256)
void k_ln2(const float* __restrict__ hsrc, const float* __restrict__ g,
           const float* __restrict__ be, unsigned short* __restrict__ xln) {
    int row  = blockIdx.x * 8 + (threadIdx.x >> 5);
    int lane = threadIdx.x & 31;
    const float* src = hsrc + (size_t)row * CC;
    float x[16], s = 0.f, s2 = 0.f;
#pragma unroll
    for (int i = 0; i < 16; ++i) { float v = src[lane + i * 32]; x[i] = v; s += v; s2 += v * v; }
#pragma unroll
    for (int m = 1; m <= 16; m <<= 1) { s += __shfl_xor(s, m, 32); s2 += __shfl_xor(s2, m, 32); }
    float mu = s * (1.f / CC);
    float inv = rsqrtf(s2 * (1.f / CC) - mu * mu + 1e-5f);
    unsigned short* dst = xln + (size_t)row * CC;
#pragma unroll
    for (int i = 0; i < 16; ++i) {
        int c = lane + i * 32;
        dst[c] = f2b((x[i] - mu) * inv * g[c] + be[c]);
    }
}

// ---------------- tiled bf16 WMMA GEMM: out = A[M,K] * Wt[N,K]^T + bias -------
// Block tile 128x128, 8 waves, wave tile 32x64 (8 WMMAs / wave / k-step).
// EPI 0: bf16 store   1: GELU -> bf16   2: f32 out += v   3: proj scatter + shortcut
template <int EPI>
__global__ __launch_bounds__(256)
void k_gemm(const unsigned short* __restrict__ A, const unsigned short* __restrict__ Wt,
            const float* __restrict__ bias, unsigned short* __restrict__ outb,
            float* __restrict__ outf, const float* __restrict__ shortcut,
            int M, int Nc, int K) {
    __shared__ unsigned short lds_a[128 * 32];   // 8 KB
    __shared__ unsigned short lds_b[128 * 32];   // 8 KB
    int tid = threadIdx.x, lane = tid & 31, wave = tid >> 5;
    int wy = wave >> 1, wx = wave & 1;
    size_t rowBase = (size_t)blockIdx.x * 128;
    int    colBase = blockIdx.y * 128;
    v8f acc[2][4] = {};

    for (int k0 = 0; k0 < K; k0 += 32) {
#ifdef HAVE_TDM
        // A tile (128 rows x 32 cols bf16) via Tensor Data Mover, wave 0 issues.
        if (tid < 32) {
            tdm_load_2d((unsigned)(size_t)(void*)lds_a,
                        A + rowBase * K + k0, 32u, 128u, (unsigned)K);
            __builtin_amdgcn_s_wait_tensorcnt(0);
        }
#else
#pragma unroll
        for (int i = 0; i < 8; ++i) {
            int idx = i * 256 + tid;
            int r = idx >> 4, cp = idx & 15;
            ((unsigned int*)lds_a)[idx] =
                *(const unsigned int*)(A + (rowBase + r) * K + k0 + cp * 2);
        }
#endif
        // B tile (128 out-cols x 32 k) staged as dwords
#pragma unroll
        for (int i = 0; i < 8; ++i) {
            int idx = i * 256 + tid;
            int n = idx >> 4, cp = idx & 15;
            ((unsigned int*)lds_b)[idx] =
                *(const unsigned int*)(Wt + (size_t)(colBase + n) * K + k0 + cp * 2);
        }
        __syncthreads();
        if (k0 + 32 < K) {   // hint next tiles into cache (global_prefetch_b8)
            __builtin_prefetch(A + (rowBase + (tid & 127)) * K + k0 + 32, 0, 1);
            __builtin_prefetch(Wt + (size_t)(colBase + (tid & 127)) * K + k0 + 32, 0, 1);
        }
        v16bf a0 = load_a_frag(lds_a + (size_t)(wy * 32) * 32, 32);
        v16bf a1 = load_a_frag(lds_a + (size_t)(wy * 32 + 16) * 32, 32);
#pragma unroll
        for (int jt = 0; jt < 4; ++jt) {
            v16bf bj = load_b_colk(lds_b + (size_t)(wx * 64 + jt * 16) * 32, 32);
            acc[0][jt] = wmma_bf16(a0, bj, acc[0][jt]);
            acc[1][jt] = wmma_bf16(a1, bj, acc[1][jt]);
        }
        __syncthreads();
    }

    int hi = lane >> 4, ln = lane & 15;
#pragma unroll
    for (int it = 0; it < 2; ++it)
#pragma unroll
        for (int jt = 0; jt < 4; ++jt)
#pragma unroll
            for (int r = 0; r < 8; ++r) {
                size_t row = rowBase + wy * 32 + it * 16 + r + 8 * hi;
                int col = colBase + wx * 64 + jt * 16 + ln;
                float v = acc[it][jt][r] + bias[col];
                if (EPI == 0) {
                    outb[row * Nc + col] = f2b(v);
                } else if (EPI == 1) {
                    float ge = 0.5f * v * (1.0f + erff(v * 0.70710678118654752f));
                    outb[row * Nc + col] = f2b(ge);
                } else if (EPI == 2) {
                    outf[row * Nc + col] += v;
                } else {        // proj: window-reverse + roll(+SS) + shortcut add
                    int rr = (int)row;
                    int win = rr >> 6, tok = rr & 63;
                    int b = win >> 6, wyy = (win >> 3) & 7, wxx = win & 7;
                    int iy = tok >> 3, ix = tok & 7;
                    int h2 = (wyy * WSZ + iy + SSH) & 63;
                    int w2 = (wxx * WSZ + ix + SSH) & 63;
                    size_t oidx = (((size_t)b * 4096) + (size_t)h2 * 64 + w2) * CC + col;
                    outf[oidx] = shortcut[oidx] + v;
                }
            }
}

// ---------------- attention core: per (window, head) ----------------
__device__ __forceinline__ int region_id(int x) { return (x < 56) ? 0 : ((x < 60) ? 1 : 2); }

__global__ __launch_bounds__(128)
void k_attn(const unsigned short* __restrict__ qb, const unsigned short* __restrict__ kb,
            const unsigned short* __restrict__ vb, const float* __restrict__ btab,
            unsigned short* __restrict__ ctx) {
    int win = blockIdx.x, head = blockIdx.y;
    __shared__ unsigned short lds_k[64 * 32];
    __shared__ unsigned short lds_v[64 * 32];
    __shared__ unsigned short lds_p[4][16 * 64];
    int tid = threadIdx.x, lane = tid & 31, wave = tid >> 5;

    const unsigned short* ksrc = kb + ((size_t)win * NTOK) * CC + head * HDIM;
    const unsigned short* vsrc = vb + ((size_t)win * NTOK) * CC + head * HDIM;
#pragma unroll
    for (int i = 0; i < 16; ++i) {
        int idx = i * 128 + tid;
        int n = idx >> 5, d = idx & 31;
        lds_k[idx] = ksrc[(size_t)n * CC + d];
        lds_v[idx] = vsrc[(size_t)n * CC + d];
    }
    __syncthreads();

    const unsigned short* qsrc = qb + ((size_t)win * NTOK + wave * 16) * CC + head * HDIM;
    v16bf aq = load_a_frag(qsrc, CC);
    v8f zero = {};
    v8f s[4];
#pragma unroll
    for (int t = 0; t < 4; ++t)
        s[t] = wmma_bf16(aq, load_b_colk(lds_k + (size_t)(t * 16) * 32, 32), zero);

    int hi = lane >> 4, ln = lane & 15;
    int wy = (win >> 3) & 7, wx = win & 7;
    const float sc = 0.17677669529663687f;        // 1/sqrt(32)
    float sv[4][8];
#pragma unroll
    for (int t = 0; t < 4; ++t)
#pragma unroll
        for (int r = 0; r < 8; ++r) {
            int i = wave * 16 + r + 8 * hi;
            int j = t * 16 + ln;
            int iy = i >> 3, ix = i & 7, jy = j >> 3, jx = j & 7;
            int bidx = (iy - jy + 7) * 15 + (ix - jx + 7);
            float bv = btab[bidx * NHEAD + head];
            int idi = region_id(wy * WSZ + iy) * 3 + region_id(wx * WSZ + ix);
            int idj = region_id(wy * WSZ + jy) * 3 + region_id(wx * WSZ + jx);
            float mv = (idi == idj) ? 0.f : -100.f;
            sv[t][r] = s[t][r] * sc + bv + mv;
        }

#pragma unroll
    for (int r = 0; r < 8; ++r) {
        float m = sv[0][r];
#pragma unroll
        for (int t = 1; t < 4; ++t) m = fmaxf(m, sv[t][r]);
#pragma unroll
        for (int msk = 1; msk <= 8; msk <<= 1) m = fmaxf(m, __shfl_xor(m, msk, 32));
        float sum = 0.f;
#pragma unroll
        for (int t = 0; t < 4; ++t) { float e = __expf(sv[t][r] - m); sv[t][r] = e; sum += e; }
#pragma unroll
        for (int msk = 1; msk <= 8; msk <<= 1) sum += __shfl_xor(sum, msk, 32);
        float inv = 1.f / sum;
#pragma unroll
        for (int t = 0; t < 4; ++t) sv[t][r] *= inv;
    }

#pragma unroll
    for (int t = 0; t < 4; ++t)
#pragma unroll
        for (int r = 0; r < 8; ++r) {
            int row = r + 8 * hi;
            lds_p[wave][row * 64 + t * 16 + ln] = f2b(sv[t][r]);
        }
    __syncthreads();

    v8f c0 = {}, c1 = {};
#pragma unroll
    for (int kt = 0; kt < 2; ++kt) {
        v16bf ap = load_a_frag(&lds_p[wave][kt * 32], 64);
        c0 = wmma_bf16(ap, load_b_rowk(lds_v + (size_t)(kt * 32) * 32 + 0, 32), c0);
        c1 = wmma_bf16(ap, load_b_rowk(lds_v + (size_t)(kt * 32) * 32 + 16, 32), c1);
    }
    unsigned short* cdst = ctx + ((size_t)win * NTOK + wave * 16) * CC + head * HDIM;
#pragma unroll
    for (int r = 0; r < 8; ++r) {
        int row = r + 8 * hi;
        cdst[(size_t)row * CC + ln]      = f2b(c0[r]);
        cdst[(size_t)row * CC + 16 + ln] = f2b(c1[r]);
    }
}

// ---------------- host-side orchestration ----------------
extern "C" void kernel_launch(void* const* d_in, const int* in_sizes, int n_in,
                              void* d_out, int out_size, void* d_ws, size_t ws_size,
                              hipStream_t stream) {
    const float* hs    = (const float*)d_in[0];
    const float* q_w   = (const float*)d_in[1];
    const float* q_b   = (const float*)d_in[2];
    const float* k_w   = (const float*)d_in[3];
    const float* k_b   = (const float*)d_in[4];
    const float* v_w   = (const float*)d_in[5];
    const float* v_b   = (const float*)d_in[6];
    const float* p_w   = (const float*)d_in[7];
    const float* p_b   = (const float*)d_in[8];
    const float* btab  = (const float*)d_in[9];
    const float* ln1w  = (const float*)d_in[10];
    const float* ln1b  = (const float*)d_in[11];
    const float* ln2w  = (const float*)d_in[12];
    const float* ln2b  = (const float*)d_in[13];
    const float* fc1w  = (const float*)d_in[14];
    const float* fc1b  = (const float*)d_in[15];
    const float* fc2w  = (const float*)d_in[16];
    const float* fc2b  = (const float*)d_in[17];
    unsigned short* W = (unsigned short*)d_ws;
    float* out = (float*)d_out;

    k_cvt<<<(262144 + 255) / 256, 256, 0, stream>>>(q_w,  W + OFF_WQ, 262144);
    k_cvt<<<(262144 + 255) / 256, 256, 0, stream>>>(k_w,  W + OFF_WK, 262144);
    k_cvt<<<(262144 + 255) / 256, 256, 0, stream>>>(v_w,  W + OFF_WV, 262144);
    k_cvt<<<(262144 + 255) / 256, 256, 0, stream>>>(p_w,  W + OFF_WP, 262144);
    k_cvt<<<(1048576 + 255) / 256, 256, 0, stream>>>(fc1w, W + OFF_W1, 1048576);
    k_cvt<<<(1048576 + 255) / 256, 256, 0, stream>>>(fc2w, W + OFF_W2, 1048576);

    k_prep<<<ROWS / 8, 256, 0, stream>>>(hs, ln1w, ln1b, W + OFF_XW);

    dim3 g1(ROWS / 128, CC / 128);
    k_gemm<0><<<g1, 256, 0, stream>>>(W + OFF_XW, W + OFF_WQ, q_b, W + OFF_Q, nullptr, nullptr, ROWS, CC, CC);
    k_gemm<0><<<g1, 256, 0, stream>>>(W + OFF_XW, W + OFF_WK, k_b, W + OFF_K, nullptr, nullptr, ROWS, CC, CC);
    k_gemm<0><<<g1, 256, 0, stream>>>(W + OFF_XW, W + OFF_WV, v_b, W + OFF_V, nullptr, nullptr, ROWS, CC, CC);

    k_attn<<<dim3(NWIN, NHEAD), 128, 0, stream>>>(W + OFF_Q, W + OFF_K, W + OFF_V, btab, W + OFF_CTX);

    k_gemm<3><<<g1, 256, 0, stream>>>(W + OFF_CTX, W + OFF_WP, p_b, nullptr, out, hs, ROWS, CC, CC);

    k_ln2<<<ROWS / 8, 256, 0, stream>>>(out, ln2w, ln2b, W + OFF_XLN);
    dim3 g2(ROWS / 128, MLPD / 128);
    k_gemm<1><<<g2, 256, 0, stream>>>(W + OFF_XLN, W + OFF_W1, fc1b, W + OFF_Y1, nullptr, nullptr, ROWS, MLPD, CC);
    k_gemm<2><<<g1, 256, 0, stream>>>(W + OFF_Y1, W + OFF_W2, fc2b, nullptr, out, nullptr, ROWS, CC, MLPD);
}
